// LEMURS_actor_85272280695435
// MI455X (gfx1250) — matrix operand order repaired and checked by reading, not compiled, not run on hardware
//
#include <hip/hip_runtime.h>

typedef __attribute__((ext_vector_type(16))) _Float16 v16h;
typedef __attribute__((ext_vector_type(8)))  _Float16 v8h;
typedef __attribute__((ext_vector_type(8)))  float    v8f;

#define WAVES 2
#define H2 128
#define H1 64

__device__ __forceinline__ float fast_rcp(float x) { return __builtin_amdgcn_rcpf(x); }
__device__ __forceinline__ float silu_f(float z) {
    // z * sigmoid(z) with native v_rcp_f32 (avoids IEEE div expansion)
    return z * fast_rcp(1.f + __expf(-z));
}

// ---- A fragment (16x32 f16) from f16 LDS row-major [16][K], k-block kb ----
// Layout (contiguous K, same order as B loaders): lanes 0-15 = rows M, hg=lane>>4 selects K sub-halves:
// e=0..15 -> K = kb*32 + hg*16 + e
// Both 8-half runs are 16B aligned -> two ds_load_b128.
__device__ __forceinline__ v16h load_a_lds(const _Float16* src, int K, int lane, int kb) {
    const int m = lane & 15, hg = lane >> 4;
    const _Float16* p = src + m * K + kb * 32 + hg * 16;
    v8h lo = *(const v8h*)(p);
    v8h hi = *(const v8h*)(p + 8);
    return __builtin_shufflevector(lo, hi, 0, 1, 2, 3, 4, 5, 6, 7,
                                           8, 9, 10, 11, 12, 13, 14, 15);
}

// ---- B fragment (32x16 f16) from f32 global row-major W[nmax][Ktot] ----
// Layout: lane&15 = N column, hg picks K half; 16 consecutive K per lane (64B).
__device__ __forceinline__ v16h load_b_glb(const float* __restrict__ W, int Ktot,
                                           int ntile, int kb, int lane) {
    const int n  = ntile * 16 + (lane & 15);
    const int hg = lane >> 4;
    const float4* p = (const float4*)(W + n * Ktot + kb * 32 + hg * 16);
    float4 f0 = p[0], f1 = p[1], f2 = p[2], f3 = p[3];
    v16h b;
    b[0]  = (_Float16)f0.x; b[1]  = (_Float16)f0.y; b[2]  = (_Float16)f0.z; b[3]  = (_Float16)f0.w;
    b[4]  = (_Float16)f1.x; b[5]  = (_Float16)f1.y; b[6]  = (_Float16)f1.z; b[7]  = (_Float16)f1.w;
    b[8]  = (_Float16)f2.x; b[9]  = (_Float16)f2.y; b[10] = (_Float16)f2.z; b[11] = (_Float16)f2.w;
    b[12] = (_Float16)f3.x; b[13] = (_Float16)f3.y; b[14] = (_Float16)f3.z; b[15] = (_Float16)f3.w;
    return b;
}

// B fragment with N guard and K zero-pad (scalar path: only small matrices)
__device__ __forceinline__ v16h load_b_glb_guard(const float* __restrict__ W, int Ktot,
                                                 int ntile, int kb, int lane, int nmax) {
    const int n  = ntile * 16 + (lane & 15);
    const int hg = lane >> 4;
    v16h b;
#pragma unroll
    for (int e = 0; e < 16; ++e) {
        const int k = kb * 32 + hg * 16 + e;
        float val = 0.f;
        if (n < nmax && k < Ktot) val = W[n * Ktot + k];
        b[e] = (_Float16)val;
    }
    return b;
}

// A fragment from f32 global X[B][Ktot], K zero-padded to 32
__device__ __forceinline__ v16h load_a_glb_kpad(const float* __restrict__ X, int Ktot,
                                                int row0, int lane) {
    const int m = lane & 15, hg = lane >> 4;
    v16h a;
#pragma unroll
    for (int e = 0; e < 8; ++e) {
        const int k0 = hg * 16 + e;
        const int k1 = hg * 16 + 8 + e;
        a[e]     = (k0 < Ktot) ? (_Float16)X[(row0 + m) * Ktot + k0] : (_Float16)0.f;
        a[e + 8] = (k1 < Ktot) ? (_Float16)X[(row0 + m) * Ktot + k1] : (_Float16)0.f;
    }
    return a;
}

// ---- C tile stores (ISA: lane&15 = N, VGPR r -> M = (lane>>4)*8 + r), fused bias+silu ----
__device__ __forceinline__ void store_c_silu_f32(v8f c, float* dst, int stride, int ntile,
                                                 const float* __restrict__ bias, int nmax, int lane) {
    const int n  = ntile * 16 + (lane & 15);
    const int mb = (lane >> 4) * 8;
    const float bv = (n < nmax) ? bias[n] : 0.f;
#pragma unroll
    for (int r = 0; r < 8; ++r)
        dst[(mb + r) * stride + n] = silu_f(c[r] + bv);
}

__device__ __forceinline__ void store_c_silu_f16(v8f c, _Float16* dst, int stride, int ntile,
                                                 const float* __restrict__ bias, int nmax, int lane) {
    const int n  = ntile * 16 + (lane & 15);
    const int mb = (lane >> 4) * 8;
    const float bv = (n < nmax) ? bias[n] : 0.f;
#pragma unroll
    for (int r = 0; r < 8; ++r)
        dst[(mb + r) * stride + n] = (_Float16)silu_f(c[r] + bv);
}

#define WMMA(c, a, b) \
    (c) = __builtin_amdgcn_wmma_f32_16x16x32_f16(false, (a), false, (b), (short)0, (c), false, false)

__global__ __launch_bounds__(WAVES * 32)
void LEMURS_fused_mlp_attn_kernel(
    const float* __restrict__ x,
    const float* __restrict__ W_in, const float* __restrict__ b_in,
    const float* __restrict__ Aq4, const float* __restrict__ Bq4,
    const float* __restrict__ Ak4, const float* __restrict__ Bk4,
    const float* __restrict__ Av4, const float* __restrict__ Bv4,
    const float* __restrict__ W_h, const float* __restrict__ b_h,
    const float* __restrict__ Aq7, const float* __restrict__ Bq7,
    const float* __restrict__ Ak7, const float* __restrict__ Bk7,
    const float* __restrict__ Av7, const float* __restrict__ Bv7,
    const float* __restrict__ W_out, const float* __restrict__ b_out,
    float* __restrict__ out)
{
    __shared__ _Float16 sA[WAVES][16 * 192];      // per-wave f16 A-source (h0 / attn / h1)
    __shared__ float    sQ[WAVES][16 * H2];
    __shared__ float    sK[WAVES][16 * H2];
    __shared__ float    sV[WAVES][16 * H2];
    __shared__ float    sMx[WAVES][16], sMn[WAVES][16];

    const int lane = threadIdx.x & 31;
    const int wv   = threadIdx.x >> 5;
    const int row0 = (blockIdx.x * WAVES + wv) * 16;

    _Float16* bufA = &sA[wv][0];            // 16x128 region
    _Float16* bufB = &sA[wv][16 * H2];      // 16x64 region
    float* q = &sQ[wv][0];
    float* k = &sK[wv][0];
    float* v = &sV[wv][0];

    // Warm L2 for the big weight matrices (global_prefetch_b8 path)
    __builtin_prefetch(Aq4, 0, 2);
    __builtin_prefetch(Ak4, 0, 2);
    __builtin_prefetch(Av4, 0, 2);

    // ---- Stage 1: h0 = silu(x @ W_in^T + b_in)  (16x128, K=12 padded to 32) ----
    {
        v16h a = load_a_glb_kpad(x, 12, row0, lane);
#pragma unroll
        for (int nt = 0; nt < 8; ++nt) {
            v8f c = {};
            v16h b = load_b_glb_guard(W_in, 12, nt, 0, lane, H2);
            WMMA(c, a, b);
            store_c_silu_f16(c, bufA, H2, nt, b_in, H2, lane);
        }
    }
    __syncthreads();

    // ---- Stage 2: q,k,v = silu(h0 @ A^T + B)  (16x128 each, K=128) ----
    {
        v16h af[4];
#pragma unroll
        for (int kb = 0; kb < 4; ++kb) af[kb] = load_a_lds(bufA, H2, lane, kb);
        const float* Ws[3] = {Aq4, Ak4, Av4};
        const float* Bs[3] = {Bq4, Bk4, Bv4};
        float*       Os[3] = {q, k, v};
#pragma unroll
        for (int g = 0; g < 3; ++g) {
            for (int nt = 0; nt < 8; ++nt) {
                v8f c = {};
#pragma unroll
                for (int kb = 0; kb < 4; ++kb) {
                    v16h b = load_b_glb(Ws[g], H2, nt, kb, lane);
                    WMMA(c, af[kb], b);
                }
                store_c_silu_f32(c, Os[g], H2, nt, Bs[g], H2, lane);
            }
        }
    }
    __syncthreads();

    // per-row k min/max for exact softmax shift: max_j q_i*k_j = q_i*kmax or q_i*kmin
    if (lane < 16) {
        float mx = -1e30f, mn = 1e30f;
        for (int j = 0; j < H2; j += 4) {
            float4 t = *(const float4*)&k[lane * H2 + j];
            mx = fmaxf(mx, fmaxf(fmaxf(t.x, t.y), fmaxf(t.z, t.w)));
            mn = fminf(mn, fminf(fminf(t.x, t.y), fminf(t.z, t.w)));
        }
        sMx[wv][lane] = mx; sMn[wv][lane] = mn;
    }
    __syncthreads();

    // ---- Stage 3: attn4 -> bufA f16 (16x128) ----
    for (int t = 0; t < 16 * H2 / 32; ++t) {
        const int p = t * 32 + lane;
        const int m = p >> 7;          // / H2
        const int i = p & (H2 - 1);
        const float qi = q[m * H2 + i];
        const float c0 = (qi > 0.f) ? qi * sMx[wv][m] : qi * sMn[wv][m];
        float Z = 0.f, acc = 0.f;
        const float* kr = &k[m * H2];
        const float* vr = &v[m * H2];
        for (int j = 0; j < H2; j += 4) {
            float4 kk = *(const float4*)&kr[j];
            float4 vv = *(const float4*)&vr[j];
            float e0 = __expf(qi * kk.x - c0);
            float e1 = __expf(qi * kk.y - c0);
            float e2 = __expf(qi * kk.z - c0);
            float e3 = __expf(qi * kk.w - c0);
            Z   += (e0 + e1) + (e2 + e3);
            acc += e0 * vv.x + e1 * vv.y + e2 * vv.z + e3 * vv.w;
        }
        bufA[m * H2 + i] = (_Float16)silu_f(acc * fast_rcp(Z));
    }
    __syncthreads();

    // ---- Stage 4: h1 = silu(attn4 @ W_h^T + b_h) -> bufB f16 (16x64, K=128) ----
    {
        v16h af[4];
#pragma unroll
        for (int kb = 0; kb < 4; ++kb) af[kb] = load_a_lds(bufA, H2, lane, kb);
#pragma unroll
        for (int nt = 0; nt < 4; ++nt) {
            v8f c = {};
#pragma unroll
            for (int kb = 0; kb < 4; ++kb) {
                v16h b = load_b_glb(W_h, H2, nt, kb, lane);
                WMMA(c, af[kb], b);
            }
            store_c_silu_f16(c, bufB, H1, nt, b_h, H1, lane);
        }
    }
    __syncthreads();

    // ---- Stage 5: q7,k7,v7 = silu(h1 @ A^T + B)  (16x64 each, K=64) ----
    {
        v16h af[2];
#pragma unroll
        for (int kb = 0; kb < 2; ++kb) af[kb] = load_a_lds(bufB, H1, lane, kb);
        const float* Ws[3] = {Aq7, Ak7, Av7};
        const float* Bs[3] = {Bq7, Bk7, Bv7};
        float*       Os[3] = {q, k, v};
#pragma unroll
        for (int g = 0; g < 3; ++g) {
            for (int nt = 0; nt < 4; ++nt) {
                v8f c = {};
#pragma unroll
                for (int kb = 0; kb < 2; ++kb) {
                    v16h b = load_b_glb(Ws[g], H1, nt, kb, lane);
                    WMMA(c, af[kb], b);
                }
                store_c_silu_f32(c, Os[g], H1, nt, Bs[g], H1, lane);
            }
        }
    }
    __syncthreads();

    if (lane < 16) {
        float mx = -1e30f, mn = 1e30f;
        for (int j = 0; j < H1; j += 4) {
            float4 t = *(const float4*)&k[lane * H1 + j];
            mx = fmaxf(mx, fmaxf(fmaxf(t.x, t.y), fmaxf(t.z, t.w)));
            mn = fminf(mn, fminf(fminf(t.x, t.y), fminf(t.z, t.w)));
        }
        sMx[wv][lane] = mx; sMn[wv][lane] = mn;
    }
    __syncthreads();

    // ---- Stage 6: attn7 -> bufA f16 (16x64) ----
    for (int t = 0; t < 16 * H1 / 32; ++t) {
        const int p = t * 32 + lane;
        const int m = p >> 6;          // / H1
        const int i = p & (H1 - 1);
        const float qi = q[m * H1 + i];
        const float c0 = (qi > 0.f) ? qi * sMx[wv][m] : qi * sMn[wv][m];
        float Z = 0.f, acc = 0.f;
        const float* kr = &k[m * H1];
        const float* vr = &v[m * H1];
        for (int j = 0; j < H1; j += 4) {
            float4 kk = *(const float4*)&kr[j];
            float4 vv = *(const float4*)&vr[j];
            float e0 = __expf(qi * kk.x - c0);
            float e1 = __expf(qi * kk.y - c0);
            float e2 = __expf(qi * kk.z - c0);
            float e3 = __expf(qi * kk.w - c0);
            Z   += (e0 + e1) + (e2 + e3);
            acc += e0 * vv.x + e1 * vv.y + e2 * vv.z + e3 * vv.w;
        }
        bufA[m * H1 + i] = (_Float16)silu_f(acc * fast_rcp(Z));
    }
    __syncthreads();

    // ---- Stage 7: y = silu(h2 @ W_out^T + b_out)  (16x25, K=64) -> q LDS [16][32] ----
    {
        v16h af[2];
#pragma unroll
        for (int kb = 0; kb < 2; ++kb) af[kb] = load_a_lds(bufA, H1, lane, kb);
#pragma unroll
        for (int nt = 0; nt < 2; ++nt) {
            v8f c = {};
#pragma unroll
            for (int kb = 0; kb < 2; ++kb) {
                v16h b = load_b_glb_guard(W_out, H1, nt, kb, lane, 25);
                WMMA(c, af[kb], b);
            }
            store_c_silu_f32(c, q, 32, nt, b_out, 25, lane);
        }
    }
    __syncthreads();

    // ---- Stage 8: per-sample quadratic reduction ----
    if (lane < 16) {
        const float* y = q + lane * 32;
        float M11 = 0.f, M12 = 0.f, M21 = 0.f, M22 = 0.f, Mpp = 0.f;
#pragma unroll
        for (int t = 0; t < 5; ++t) {
            M11 += y[t] * y[t];           M12 += y[5 + t] * y[5 + t];
            M21 += y[10 + t] * y[10 + t]; M22 += y[15 + t] * y[15 + t];
            Mpp += y[20 + t] * y[20 + t];
        }
        const float q0 = y[0], q1 = y[1], q2 = y[2], q3 = y[3];
        const float quad = M11 * (q0 * q0 + q1 * q1)
                         + (M12 + M21) * (q0 * q2 + q1 * q3)
                         + M22 * (q2 * q2 + q3 * q3);
        out[row0 + lane] = quad + Mpp;
    }
}

extern "C" void kernel_launch(void* const* d_in, const int* in_sizes, int n_in,
                              void* d_out, int out_size, void* d_ws, size_t ws_size,
                              hipStream_t stream) {
    (void)in_sizes; (void)n_in; (void)d_ws; (void)ws_size;
    // setup_inputs order: x, na, W_in, b_in, Aq4, Bq4, Ak4, Bk4, Av4, Bv4,
    //                     W_h, b_h, Aq7, Bq7, Ak7, Bk7, Av7, Bv7, W_out, b_out
    const float* x     = (const float*)d_in[0];
    const float* W_in  = (const float*)d_in[2];
    const float* b_in  = (const float*)d_in[3];
    const float* Aq4   = (const float*)d_in[4];
    const float* Bq4   = (const float*)d_in[5];
    const float* Ak4   = (const float*)d_in[6];
    const float* Bk4   = (const float*)d_in[7];
    const float* Av4   = (const float*)d_in[8];
    const float* Bv4   = (const float*)d_in[9];
    const float* W_h   = (const float*)d_in[10];
    const float* b_h   = (const float*)d_in[11];
    const float* Aq7   = (const float*)d_in[12];
    const float* Bq7   = (const float*)d_in[13];
    const float* Ak7   = (const float*)d_in[14];
    const float* Bk7   = (const float*)d_in[15];
    const float* Av7   = (const float*)d_in[16];
    const float* Bv7   = (const float*)d_in[17];
    const float* W_out = (const float*)d_in[18];
    const float* b_out = (const float*)d_in[19];
    float* out = (float*)d_out;

    const int B = 16384;                       // out_size == B
    const int blocks = B / (16 * WAVES);       // 512 blocks x 64 threads
    LEMURS_fused_mlp_attn_kernel<<<blocks, WAVES * 32, 0, stream>>>(
        x, W_in, b_in, Aq4, Bq4, Ak4, Bk4, Av4, Bv4,
        W_h, b_h, Aq7, Bq7, Ak7, Bk7, Av7, Bv7, W_out, b_out, out);
    (void)out_size;
}
